// _Conv1D_68805376082214
// MI455X (gfx1250) — matrix operand order repaired and checked by reading, not compiled, not run on hardware
//
#include <hip/hip_runtime.h>

// ---------------------------------------------------------------------------
// Causal Conv1D  (B=16, S=2048, C=256, F=256, K=5)  as implicit GEMM on CDNA5
//   M = B*S = 32768, N = F = 256, Kdim = K*C = 1280
//   FLOPs ~21.5G, HBM floor ~67MB -> f16 WMMA path (fp32 accum) is the right
//   precision/throughput point on MI455X.
// ---------------------------------------------------------------------------

typedef __attribute__((ext_vector_type(16))) _Float16 v16h;
typedef __attribute__((ext_vector_type(8)))  _Float16 v8h;
typedef __attribute__((ext_vector_type(4)))  _Float16 v4h;
typedef __attribute__((ext_vector_type(8)))  float    v8f;
typedef __attribute__((ext_vector_type(4)))  float    v4f;

#define BATCH 16
#define SEQ   2048
#define CH    256
#define FILT  256
#define KW    5
#define KC    (KW * CH)          // 1280 = GEMM K dimension
#define NCHUNK (KC / 32)         // 40 k-chunks of 32 per WMMA
#define ROWS_PER_WG 64           // seq rows per workgroup
#define LDSROWS (ROWS_PER_WG + KW - 1)   // 68 staged input rows
#define LDSSTRIDE 264            // 256 + 8 f16 pad: 528B row stride = 4 dwords
                                 // mod 64 banks -> conflict-free ds_load_b128

// ---------------------------------------------------------------------------
// Prep: swizzle fp32 weights w[f][k][c] into f16 WMMA-B fragment order:
//   blob[nt(16)][chunk(40)][lane(32)][elem(16)]   (f16)
// lane = (K-half<<4) | n ,  elem i -> k_local = 16*(lane>=16) + i
// so the main kernel's B fetch is 32B/lane, fully coalesced.
// ---------------------------------------------------------------------------
__global__ __launch_bounds__(256) void conv1d_wswizzle(
    const float* __restrict__ w, _Float16* __restrict__ blob)
{
    int idx  = blockIdx.x * 256 + threadIdx.x;   // 0 .. 327679
    int e    = idx & 15;                         // element within v16h
    int lane = (idx >> 4) & 31;
    int rest = idx >> 9;
    int chunk = rest % NCHUNK;
    int nt    = rest / NCHUNK;
    int f       = nt * 16 + (lane & 15);
    int k_local = ((lane >> 4) << 4) + e;        // 0..31 within chunk
    int kc      = chunk * 32 + k_local;          // flat (k*C + c)
    blob[idx] = (_Float16)w[f * KC + kc];        // w flat [f][k][c] == [f][kc]
}

// ---------------------------------------------------------------------------
// Main: one WG = 64 seq rows x 256 filters. 8 wave32; each wave owns
// 4 M-tiles x 2 N-tiles (64x32 outputs) of v_wmma_f32_16x16x32_f16.
// ---------------------------------------------------------------------------
__global__ __launch_bounds__(256) void conv1d_wmma(
    const float* __restrict__ x,      // [B][S][C] fp32
    const _Float16* __restrict__ wh,  // swizzled f16 weights (d_ws)
    const float* __restrict__ bias,   // [F]
    float* __restrict__ out)          // [B][S][F] fp32
{
    __shared__ _Float16 xs[LDSROWS * LDSSTRIDE];

    const int batch = blockIdx.x >> 5;           // 32 row-tiles per batch
    const int s0    = (blockIdx.x & 31) << 6;    // base seq position

    const int wave = threadIdx.x >> 5;           // 0..7
    const int lane = threadIdx.x & 31;
    const int m    = lane & 15;                  // row within 16x16 tile
    const int hi   = lane >> 4;                  // K-half select
    const int n    = lane & 15;                  // col within 16x16 tile

    const int nt0 = wave * 2;                    // two N-tiles per wave
    const int nt1 = nt0 + 1;

    // B fragment base pointers into the swizzled blob (per-chunk step = 512 f16)
    const _Float16* bp0 = wh + (((size_t)nt0 * NCHUNK) * 32 + lane) * 16;
    const _Float16* bp1 = wh + (((size_t)nt1 * NCHUNK) * 32 + lane) * 16;

    // Warm L2->L0 path for the first weight chunks while we stage the input
    // (emits gfx1250 global_prefetch).
    #pragma unroll
    for (int pk = 0; pk < 4; ++pk) {
        __builtin_prefetch(bp0 + (size_t)pk * 512, 0, 0);
        __builtin_prefetch(bp1 + (size_t)pk * 512, 0, 0);
    }

    // ---- stage input window rows [s0-4, s0+63] x 256 ch into LDS as f16 ----
    // Vectorized: each thread owns a 4-channel group of one row per step:
    //   global_load_b128 (float4) -> cvt -> ds_store_b64 (4 x f16).
    {
        const int c4 = (threadIdx.x & 63) << 2;  // channel group 0,4,...,252
        const int r0 = threadIdx.x >> 6;         // starting row 0..3
        #pragma unroll 4
        for (int r = r0; r < LDSROWS; r += 4) {
            int s = s0 - (KW - 1) + r;
            v4f v = {};
            if (s >= 0)                          // causal zero-pad
                v = *(const v4f*)(x + (((size_t)batch * SEQ + s) * CH) + c4);
            v4h h; h.x = (_Float16)v.x; h.y = (_Float16)v.y;
                   h.z = (_Float16)v.z; h.w = (_Float16)v.w;
            *(v4h*)(&xs[r * LDSSTRIDE + c4]) = h;
        }
    }
    __syncthreads();

    v8f acc[4][2] = {};

    for (int ck = 0; ck < NCHUNK; ++ck) {
        const int k  = ck >> 3;                  // tap index (chunk stays in one tap)
        const int c0 = (ck & 7) << 5;            // channel base of this chunk

        union { v16h v; v8h h[2]; } b0, b1;
        const _Float16* q0 = bp0 + (size_t)ck * 512;
        const _Float16* q1 = bp1 + (size_t)ck * 512;
        b0.h[0] = *(const v8h*)(q0);
        b0.h[1] = *(const v8h*)(q0 + 8);
        b1.h[0] = *(const v8h*)(q1);
        b1.h[1] = *(const v8h*)(q1 + 8);

        #pragma unroll
        for (int mt = 0; mt < 4; ++mt) {
            // A fragment, contiguous K order (matches swizzled B):
            //   lanes 0-15: elems 0-15 -> K 0-15
            //   lanes16-31: elems 0-15 -> K 16-31
            const _Float16* ap = &xs[(mt * 16 + m + k) * LDSSTRIDE + c0 + hi * 16];
            union { v16h v; v8h h[2]; } a;
            a.h[0] = *(const v8h*)(ap);          // ds_load_b128
            a.h[1] = *(const v8h*)(ap + 8);      // ds_load_b128
            acc[mt][0] = __builtin_amdgcn_wmma_f32_16x16x32_f16(
                false, a.v, false, b0.v, (short)0, acc[mt][0], false, false);
            acc[mt][1] = __builtin_amdgcn_wmma_f32_16x16x32_f16(
                false, a.v, false, b1.v, (short)0, acc[mt][1], false, false);
        }
    }

    // ---- bias + store. C/D layout: VGPR r -> M = r + 8*hi, N = lane&15 ----
    const float bv0 = bias[nt0 * 16 + n];
    const float bv1 = bias[nt1 * 16 + n];

    #pragma unroll
    for (int mt = 0; mt < 4; ++mt) {
        float* op = out + (((size_t)batch * SEQ + s0 + mt * 16 + hi * 8) * FILT);
        #pragma unroll
        for (int r = 0; r < 8; ++r) {
            op[(size_t)r * FILT + nt0 * 16 + n] = acc[mt][0][r] + bv0;
            op[(size_t)r * FILT + nt1 * 16 + n] = acc[mt][1][r] + bv1;
        }
    }
}

// ---------------------------------------------------------------------------
extern "C" void kernel_launch(void* const* d_in, const int* in_sizes, int n_in,
                              void* d_out, int out_size, void* d_ws, size_t ws_size,
                              hipStream_t stream)
{
    const float* x    = (const float*)d_in[0];   // (16,2048,256) fp32
    const float* w    = (const float*)d_in[1];   // (256,5,256)   fp32
    const float* bias = (const float*)d_in[2];   // (256,)        fp32
    float*       out  = (float*)d_out;           // (16,2048,256) fp32
    _Float16*    wh   = (_Float16*)d_ws;         // 640 KB swizzled f16 weights

    (void)in_sizes; (void)n_in; (void)out_size; (void)ws_size;

    // 1) one-time weight f32->f16 swizzle into WMMA B-fragment order
    conv1d_wswizzle<<<(FILT * KC) / 256, 256, 0, stream>>>(w, wh);

    // 2) implicit-GEMM conv: 512 WGs x 256 threads (8 wave32 each)
    conv1d_wmma<<<(BATCH * SEQ) / ROWS_PER_WG, 256, 0, stream>>>(x, wh, bias, out);
}
